// EfficientAttention_23682449670168
// MI455X (gfx1250) — matrix operand order repaired and checked by reading, not compiled, not run on hardware
//
#include <hip/hip_runtime.h>
#include <hip/hip_bf16.h>

// Problem constants (from reference)
#define B_  4
#define S_  4096
#define DM_ 1024
#define H_  16
#define HD_ 64
#define M_  (B_ * S_)   // 16384 rows
#define K_  DM_         // 1024
#define N_  DM_         // 1024

#define KSTEP      64   // K elements staged per TDM transfer
#define LDS_STRIDE 72   // bf16 elems per LDS row: 64 data + 8 pad (16B) -> 144B row

typedef __attribute__((ext_vector_type(16))) __bf16 v16bf;
typedef __attribute__((ext_vector_type(8)))  __bf16 v8bf;
typedef __attribute__((ext_vector_type(4)))  __bf16 v4bf;
typedef __attribute__((ext_vector_type(8)))  float  v8f;
typedef __attribute__((ext_vector_type(4)))  unsigned int u32x4;
typedef __attribute__((ext_vector_type(8)))  int i32x8;
typedef __attribute__((ext_vector_type(4)))  int i32x4;

union BF16x16 { v16bf v; v8bf h[2]; };

#if defined(__has_builtin)
#if __has_builtin(__builtin_amdgcn_tensor_load_to_lds) && \
    __has_builtin(__builtin_amdgcn_s_wait_tensorcnt)
#define HAVE_TDM 1
#endif
#endif
#ifndef HAVE_TDM
#define HAVE_TDM 0
#endif

// ---------------------------------------------------------------------------
// TDM: DMA one [128 rows x 64 cols] bf16 tile (row stride = strideElems elems)
// from global memory into LDS at byte offset lds_off. TDM pads each 128B data
// row with 16B so the LDS row stride is 144B (= LDS_STRIDE bf16 elems).
// D# layout per CDNA5 ISA 8.3/8.4. tensor_dim1 fixed at 1<<20 (tiles are
// always fully in-bounds here).
// ---------------------------------------------------------------------------
#if HAVE_TDM
__device__ __forceinline__ void tdm_tile_load_128x64(
    unsigned long long gaddr, unsigned int lds_off,
    unsigned int tensorD0, unsigned int strideElems)
{
    u32x4 g0;
    g0[0] = 1u;                                   // count=1, user descriptor
    g0[1] = lds_off;                              // lds_addr (bytes)
    g0[2] = (unsigned int)gaddr;                  // global_addr[31:0]
    g0[3] = (unsigned int)((gaddr >> 32) & 0x01FFFFFFu) | (2u << 30); // hi | type=2

    i32x8 g1;
    g1[0] = (int)((1u << 16)    // data_size = 1 (2 bytes)
                | (1u << 20)    // pad_enable
                | (4u << 22)    // pad_interval code 4 -> every 16 x 8B = 128B
                | (3u << 25));  // pad_amount  code 3 -> 4 DWORDs = 16B
    g1[1] = (int)((tensorD0 & 0xFFFFu) << 16);    // abar=0 | tensor_dim0 lo16
    g1[2] = (int)(tensorD0 >> 16);                // dim0 hi16 | dim1 lo16 (=0)
    g1[3] = (int)(16u | (64u << 16));             // dim1 hi16 (dim1=1<<20) | tile_dim0=64
    g1[4] = 128;                                  // tile_dim1=128, tile_dim2=0
    g1[5] = (int)strideElems;                     // tensor_dim0_stride lo32
    g1[6] = 0;                                    // stride hi16 | dim1_stride lo16
    g1[7] = 0;

    i32x4 z4 = {0, 0, 0, 0};
#if defined(__clang_major__) && (__clang_major__ >= 23)
    i32x8 z8 = {0, 0, 0, 0, 0, 0, 0, 0};
    __builtin_amdgcn_tensor_load_to_lds(g0, g1, z4, z4, z8, 0);
#else
    __builtin_amdgcn_tensor_load_to_lds(g0, g1, z4, z4, 0);
#endif
}
#endif

// Fallback cooperative stage: 256 threads copy [128 x 64] bf16 tile to LDS
__device__ __forceinline__ void coop_stage_128x64(
    const __bf16* __restrict__ gsrc, int ldK, __bf16* __restrict__ sdst,
    int kk, int tid)
{
    const int row = tid >> 1;
    const int cs  = (tid & 1) * 32;
    const __bf16* src = gsrc + (size_t)row * ldK + kk + cs;
    __bf16* dst = sdst + row * LDS_STRIDE + cs;
#pragma unroll
    for (int q = 0; q < 4; q++)
        *(v8bf*)(dst + q * 8) = *(const v8bf*)(src + q * 8);
}

// ---------------------------------------------------------------------------
// Per-slab WMMA compute: 2x4 tiles of 16x16, K-depth KSTEP from LDS buffers.
// ---------------------------------------------------------------------------
__device__ __forceinline__ void compute_slab(
    const __bf16* __restrict__ aB, const __bf16* __restrict__ bB,
    v8f acc[2][4], int wm, int wn, int lr, int ka, int kb)
{
#pragma unroll
    for (int ks = 0; ks < KSTEP; ks += 32) {
        BF16x16 a0, a1, b[4];
        const __bf16* pa = aB + (wm * 32 + lr) * LDS_STRIDE + ks + ka;
        a0.h[0] = *(const v8bf*)pa;
        a0.h[1] = *(const v8bf*)(pa + 16);
        const __bf16* pa1 = pa + 16 * LDS_STRIDE;
        a1.h[0] = *(const v8bf*)pa1;
        a1.h[1] = *(const v8bf*)(pa1 + 16);
#pragma unroll
        for (int j = 0; j < 4; j++) {
            const __bf16* pb = bB + (wn * 64 + j * 16 + lr) * LDS_STRIDE + ks + kb;
            b[j].h[0] = *(const v8bf*)pb;
            b[j].h[1] = *(const v8bf*)(pb + 16);
        }
#pragma unroll
        for (int j = 0; j < 4; j++) {
            acc[0][j] = __builtin_amdgcn_wmma_f32_16x16x32_bf16(
                false, a0.v, false, b[j].v, (short)0, acc[0][j], false, false);
            acc[1][j] = __builtin_amdgcn_wmma_f32_16x16x32_bf16(
                false, a1.v, false, b[j].v, (short)0, acc[1][j], false, false);
        }
    }
}

// ---------------------------------------------------------------------------
// f32 -> bf16 conversion (4 elements / thread)
// ---------------------------------------------------------------------------
__global__ __launch_bounds__(256) void cvt_f32_to_bf16(
    const float* __restrict__ in, __bf16* __restrict__ out, long long n4)
{
    long long i = (long long)blockIdx.x * blockDim.x + threadIdx.x;
    if (i >= n4) return;
    float4 f = ((const float4*)in)[i];
    v4bf o;
    o[0] = (__bf16)f.x; o[1] = (__bf16)f.y;
    o[2] = (__bf16)f.z; o[3] = (__bf16)f.w;
    ((v4bf*)out)[i] = o;
}

// ---------------------------------------------------------------------------
// NT GEMM:  C[m,n] = sum_k A[m,k] * W[n,k]  (+ bias[n], optional elu+1)
// Block: 256 threads = 8 waves; block tile 128x128; wave tile 32x64 (2x4 WMMA)
// TDM double-buffered LDS staging (ping-pong fully unrolled so LDS bases are
// compile-time constants), K-step 64.
// ---------------------------------------------------------------------------
template<int ELU>
__global__ __launch_bounds__(256) void gemm_nt_bf16_wmma(
    const __bf16* __restrict__ A,
    const __bf16* __restrict__ W,
    const float*  __restrict__ bias,
    float* __restrict__ C,
    int M, int N, int K)
{
    __shared__ __attribute__((aligned(16))) __bf16 sA[2][128 * LDS_STRIDE];
    __shared__ __attribute__((aligned(16))) __bf16 sB[2][128 * LDS_STRIDE];

    const int tid  = threadIdx.x;
    const int lane = tid & 31;
    const int wid  = tid >> 5;
    const int lr   = lane & 15;   // row/col within 16x16 tile
    const int lh   = lane >> 4;   // half select (0/1)
    const int wm   = wid & 3;     // wave position along M (0..3)
    const int wn   = wid >> 2;    // wave position along N (0..1)

    const int m0 = blockIdx.x * 128 + wm * 32;
    const int n0 = blockIdx.y * 128 + wn * 64;

    const v8f vzero = {0.f, 0.f, 0.f, 0.f, 0.f, 0.f, 0.f, 0.f};
    v8f acc[2][4];
#pragma unroll
    for (int i = 0; i < 2; i++)
#pragma unroll
        for (int j = 0; j < 4; j++) acc[i][j] = vzero;

    const int ka = lh * 8;    // A frag: lanes 0-15 -> K 0..7 & 16..23
    const int kb = lh * 8;    // B frag: lanes 0-15 -> K 0..7 & 16..23

#if HAVE_TDM
    // Scalar (SGPR) predicate: only wave 0 *executes* the TDM clause.
    // TDM ignores EXEC, so an exec-masked lowering would duplicate the DMA.
    const bool issuer = (__builtin_amdgcn_readfirstlane((int)threadIdx.x) == 0);
    const unsigned long long gA =
        (unsigned long long)(uintptr_t)A + (unsigned long long)(blockIdx.x * 128) * K * 2ull;
    const unsigned long long gB =
        (unsigned long long)(uintptr_t)W + (unsigned long long)(blockIdx.y * 128) * K * 2ull;
    const unsigned int ldsA0 = (unsigned int)(uintptr_t)&sA[0][0];
    const unsigned int ldsA1 = (unsigned int)(uintptr_t)&sA[1][0];
    const unsigned int ldsB0 = (unsigned int)(uintptr_t)&sB[0][0];
    const unsigned int ldsB1 = (unsigned int)(uintptr_t)&sB[1][0];
#else
    const __bf16* Ablk = A + (size_t)blockIdx.x * 128 * K;
    const __bf16* Bblk = W + (size_t)blockIdx.y * 128 * K;
#endif

    // Prologue: stage K-slab 0 into buffer 0
#if HAVE_TDM
    if (issuer) {
        tdm_tile_load_128x64(gA, ldsA0, (unsigned)K, (unsigned)K);
        tdm_tile_load_128x64(gB, ldsB0, (unsigned)K, (unsigned)K);
    }
    __builtin_amdgcn_s_wait_tensorcnt(0);
#else
    coop_stage_128x64(Ablk, K, &sA[0][0], 0, tid);
    coop_stage_128x64(Bblk, K, &sB[0][0], 0, tid);
#endif
    __syncthreads();

    int kk = 0;
    for (;;) {
        // --- stage even slab in buf0: prefetch next into buf1, compute buf0
        if (kk + KSTEP < K) {
#if HAVE_TDM
            if (issuer) {
                tdm_tile_load_128x64(gA + (unsigned long long)(kk + KSTEP) * 2ull,
                                     ldsA1, (unsigned)K, (unsigned)K);
                tdm_tile_load_128x64(gB + (unsigned long long)(kk + KSTEP) * 2ull,
                                     ldsB1, (unsigned)K, (unsigned)K);
            }
#else
            coop_stage_128x64(Ablk, K, &sA[1][0], kk + KSTEP, tid);
            coop_stage_128x64(Bblk, K, &sB[1][0], kk + KSTEP, tid);
#endif
        }
        compute_slab(&sA[0][0], &sB[0][0], acc, wm, wn, lr, ka, kb);
#if HAVE_TDM
        __builtin_amdgcn_s_wait_tensorcnt(0);
#endif
        __syncthreads();
        kk += KSTEP;
        if (kk >= K) break;

        // --- odd slab in buf1: prefetch next into buf0, compute buf1
        if (kk + KSTEP < K) {
#if HAVE_TDM
            if (issuer) {
                tdm_tile_load_128x64(gA + (unsigned long long)(kk + KSTEP) * 2ull,
                                     ldsA0, (unsigned)K, (unsigned)K);
                tdm_tile_load_128x64(gB + (unsigned long long)(kk + KSTEP) * 2ull,
                                     ldsB0, (unsigned)K, (unsigned)K);
            }
#else
            coop_stage_128x64(Ablk, K, &sA[0][0], kk + KSTEP, tid);
            coop_stage_128x64(Bblk, K, &sB[0][0], kk + KSTEP, tid);
#endif
        }
        compute_slab(&sA[1][0], &sB[1][0], acc, wm, wn, lr, ka, kb);
#if HAVE_TDM
        __builtin_amdgcn_s_wait_tensorcnt(0);
#endif
        __syncthreads();
        kk += KSTEP;
        if (kk >= K) break;
    }

    // Epilogue: lane holds col n0+j*16+lr, rows m0+i*16+lh*8+r
#pragma unroll
    for (int j = 0; j < 4; j++) {
        const int n  = n0 + j * 16 + lr;
        const float bv = bias[n];
#pragma unroll
        for (int i = 0; i < 2; i++) {
            const int mb = m0 + i * 16 + lh * 8;
            float* cp = C + (size_t)mb * N + n;
#pragma unroll
            for (int r = 0; r < 8; r++) {
                float v = acc[i][j][r] + bv;
                if (ELU) v = (v > 0.f) ? (v + 1.f) : __expf(v);
                cp[(size_t)r * N] = v;
            }
        }
    }
}

// ---------------------------------------------------------------------------
// kv[d,e] = sum_s k[s,d]*v[s,e] ; ksum[d] = sum_s k[s,d]   per (b,h)
// ---------------------------------------------------------------------------
__global__ __launch_bounds__(256) void kv_ksum_kernel(
    const float* __restrict__ Kf, const float* __restrict__ Vf,
    float* __restrict__ kvb, float* __restrict__ ksumb)
{
    __shared__ float kt[64][65];
    __shared__ float vt[64][65];

    const int bh = blockIdx.x;
    const int b = bh / H_, h = bh % H_;
    const size_t base = (size_t)b * S_ * DM_ + (size_t)h * HD_;

    const int t   = threadIdx.x;
    const int r   = t >> 2;
    const int seg = t & 3;
    const int d0  = r;
    const int e0  = seg * 16;

    float acc[16];
#pragma unroll
    for (int e = 0; e < 16; e++) acc[e] = 0.f;
    float ks = 0.f;

    for (int s0 = 0; s0 < S_; s0 += 64) {
        const size_t row = base + (size_t)(s0 + r) * DM_;
#pragma unroll
        for (int q = 0; q < 4; q++) {
            const int c = seg * 16 + q * 4;
            float4 kf = *(const float4*)(Kf + row + c);
            float4 vf = *(const float4*)(Vf + row + c);
            kt[r][c] = kf.x; kt[r][c+1] = kf.y; kt[r][c+2] = kf.z; kt[r][c+3] = kf.w;
            vt[r][c] = vf.x; vt[r][c+1] = vf.y; vt[r][c+2] = vf.z; vt[r][c+3] = vf.w;
        }
        __syncthreads();
#pragma unroll 4
        for (int si = 0; si < 64; si++) {
            const float kd = kt[si][d0];
            const float* vr = &vt[si][e0];
#pragma unroll
            for (int e = 0; e < 16; e++) acc[e] += kd * vr[e];
            if (seg == 0) ks += kd;
        }
        __syncthreads();
    }

    float* kvout = kvb + (size_t)bh * (HD_ * HD_) + (size_t)d0 * HD_ + e0;
#pragma unroll
    for (int e = 0; e < 16; e++) kvout[e] = acc[e];
    if (seg == 0) ksumb[(size_t)bh * HD_ + d0] = ks;
}

// ---------------------------------------------------------------------------
// out[s,e] = (sum_d q[s,d]*kv[d,e]) / (sum_d q[s,d]*ksum[d] + 1e-6) -> bf16
// ---------------------------------------------------------------------------
__global__ __launch_bounds__(256) void qkv_norm_kernel(
    const float* __restrict__ Qf, const float* __restrict__ kvb,
    const float* __restrict__ ksumb, __bf16* __restrict__ attn)
{
    __shared__ float kvs[64][65];
    __shared__ float qt[64][65];
    __shared__ float kss[64];

    const int bh = blockIdx.x;
    const int b = bh / H_, h = bh % H_;
    const int s0 = blockIdx.y * 64;

    const int t   = threadIdx.x;
    const int r   = t >> 2;
    const int seg = t & 3;

    {
        const float* kvg = kvb + (size_t)bh * (HD_ * HD_);
#pragma unroll
        for (int q = 0; q < 4; q++) {
            const int c = seg * 16 + q * 4;
            float4 f = *(const float4*)(kvg + (size_t)r * HD_ + c);
            kvs[r][c] = f.x; kvs[r][c+1] = f.y; kvs[r][c+2] = f.z; kvs[r][c+3] = f.w;
        }
        if (t < 64) kss[t] = ksumb[(size_t)bh * HD_ + t];
    }
    const size_t base = (size_t)b * S_ * DM_ + (size_t)h * HD_;
    {
        const size_t row = base + (size_t)(s0 + r) * DM_;
#pragma unroll
        for (int q = 0; q < 4; q++) {
            const int c = seg * 16 + q * 4;
            float4 f = *(const float4*)(Qf + row + c);
            qt[r][c] = f.x; qt[r][c+1] = f.y; qt[r][c+2] = f.z; qt[r][c+3] = f.w;
        }
    }
    __syncthreads();

    const int sl = r, e0 = seg * 16;
    float nz = 1e-6f;
    float acc[16];
#pragma unroll
    for (int e = 0; e < 16; e++) acc[e] = 0.f;

#pragma unroll 4
    for (int d = 0; d < 64; d++) {
        const float qv = qt[sl][d];
        nz += qv * kss[d];
        const float* kr = &kvs[d][e0];
#pragma unroll
        for (int e = 0; e < 16; e++) acc[e] += qv * kr[e];
    }
    const float inv = 1.f / nz;

    __bf16* op = attn + (size_t)(b * S_ + s0 + sl) * DM_ + h * HD_ + e0;
    v8bf o0, o1;
#pragma unroll
    for (int e = 0; e < 8; e++) {
        o0[e] = (__bf16)(acc[e] * inv);
        o1[e] = (__bf16)(acc[8 + e] * inv);
    }
    *(v8bf*)op = o0;
    *(v8bf*)(op + 8) = o1;
}

// ---------------------------------------------------------------------------
// Host-side launcher
// ---------------------------------------------------------------------------
extern "C" void kernel_launch(void* const* d_in, const int* in_sizes, int n_in,
                              void* d_out, int out_size, void* d_ws, size_t ws_size,
                              hipStream_t stream)
{
    const float* x  = (const float*)d_in[0];
    const float* Wq = (const float*)d_in[1];
    const float* bq = (const float*)d_in[2];
    const float* Wk = (const float*)d_in[3];
    const float* bk = (const float*)d_in[4];
    const float* Wv = (const float*)d_in[5];
    const float* bv = (const float*)d_in[6];
    const float* Wo = (const float*)d_in[7];
    const float* bo = (const float*)d_in[8];
    float* out = (float*)d_out;

    char* ws = (char*)d_ws;
    size_t off = 0;
    auto alloc = [&](size_t bytes) -> void* {
        void* p = ws + off;
        off += (bytes + 255) & ~(size_t)255;
        return p;
    };

    const size_t MK = (size_t)M_ * K_;
    const size_t NK = (size_t)N_ * K_;

    __bf16* x_bf  = (__bf16*)alloc(MK * 2);   // reused as attn_bf later
    __bf16* wq_bf = (__bf16*)alloc(NK * 2);
    __bf16* wk_bf = (__bf16*)alloc(NK * 2);
    __bf16* wv_bf = (__bf16*)alloc(NK * 2);
    __bf16* wo_bf = (__bf16*)alloc(NK * 2);
    float*  Qf    = (float*)alloc(MK * 4);
    float*  Kf    = (float*)alloc(MK * 4);
    float*  Vf    = (float*)alloc(MK * 4);
    float*  kvb   = (float*)alloc((size_t)B_ * H_ * HD_ * HD_ * 4);
    float*  ksb   = (float*)alloc((size_t)B_ * H_ * HD_ * 4);
    __bf16* attn_bf = x_bf;

    {
        long long n4 = (long long)(MK / 4);
        cvt_f32_to_bf16<<<dim3((unsigned)((n4 + 255) / 256)), dim3(256), 0, stream>>>(x, x_bf, n4);
        long long w4 = (long long)(NK / 4);
        dim3 gw((unsigned)((w4 + 255) / 256));
        cvt_f32_to_bf16<<<gw, dim3(256), 0, stream>>>(Wq, wq_bf, w4);
        cvt_f32_to_bf16<<<gw, dim3(256), 0, stream>>>(Wk, wk_bf, w4);
        cvt_f32_to_bf16<<<gw, dim3(256), 0, stream>>>(Wv, wv_bf, w4);
        cvt_f32_to_bf16<<<gw, dim3(256), 0, stream>>>(Wo, wo_bf, w4);
    }

    dim3 gg(M_ / 128, N_ / 128);
    gemm_nt_bf16_wmma<1><<<gg, dim3(256), 0, stream>>>(x_bf, wq_bf, bq, Qf, M_, N_, K_);
    gemm_nt_bf16_wmma<1><<<gg, dim3(256), 0, stream>>>(x_bf, wk_bf, bk, Kf, M_, N_, K_);
    gemm_nt_bf16_wmma<0><<<gg, dim3(256), 0, stream>>>(x_bf, wv_bf, bv, Vf, M_, N_, K_);

    kv_ksum_kernel<<<dim3(B_ * H_), dim3(256), 0, stream>>>(Kf, Vf, kvb, ksb);
    qkv_norm_kernel<<<dim3(B_ * H_, S_ / 64), dim3(256), 0, stream>>>(Qf, kvb, ksb, attn_bf);

    gemm_nt_bf16_wmma<0><<<gg, dim3(256), 0, stream>>>(attn_bf, wo_bf, bo, out, M_, N_, K_);
}
